// DCAMA_model_20753281974742
// MI455X (gfx1250) — matrix operand order repaired and checked by reading, not compiled, not run on hardware
//
#include <hip/hip_runtime.h>
#include <math.h>

typedef __attribute__((ext_vector_type(16))) _Float16 v16h;
typedef __attribute__((ext_vector_type(8)))  _Float16 v8h;
typedef __attribute__((ext_vector_type(8)))  float    v8f;

// ---------------------------------------------------------------------------
// WMMA helper: D = A(16x32 f16) x B(32x16 f16) + C(16x16 f32)
// ---------------------------------------------------------------------------
__device__ __forceinline__ v8f wmma_f16(v16h a, v16h b, v8f c) {
  return __builtin_amdgcn_wmma_f32_16x16x32_f16(false, a, false, b, (short)0, c,
                                                false, false);
}

// Assemble a v16h from two aligned 8-half chunks.
__device__ __forceinline__ v16h load2x8(const _Float16* p0, const _Float16* p1) {
  v8h lo = *(const v8h*)p0;
  v8h hi = *(const v8h*)p1;
  v16h r;
#pragma unroll
  for (int t = 0; t < 8; ++t) { r[t] = lo[t]; r[t + 8] = hi[t]; }
  return r;
}

// ---------------------------------------------------------------------------
// f32 -> f16 convert (weight pre-conversion)
// ---------------------------------------------------------------------------
__global__ void k_f32_to_f16(const float* __restrict__ in,
                             _Float16* __restrict__ out, int n) {
  int i = blockIdx.x * blockDim.x + threadIdx.x;
  if (i < n) out[i] = (_Float16)in[i];
}

// ---------------------------------------------------------------------------
// Build Q/S matrices: out[l,c] = feat[c,l] + PE(l,c), converted to f16.
// ---------------------------------------------------------------------------
__global__ void k_prep_qs(const float* __restrict__ feat,
                          _Float16* __restrict__ out, int C, int L) {
  int idx = blockIdx.x * blockDim.x + threadIdx.x;
  if (idx >= L * C) return;
  int c = idx % C;
  int l = idx / C;
  // div = exp(-(2k) * ln(10000)/C), k = c/2 ; pe = even? sin : cos
  float ex  = -(float)(c & ~1) * (9.210340371976184f / (float)C);
  float ang = (float)l * __expf(ex);
  float pe  = (c & 1) ? __cosf(ang) : __sinf(ang);
  out[idx] = (_Float16)(feat[(size_t)c * L + l] + pe);
}

// ---------------------------------------------------------------------------
// GEMM: out(L x N, f16) = A(L x K, f16) @ W(N x K, f16)^T + bias(N, f32)
// One wave computes one 16x16 tile; K-loop in steps of 32 over WMMA.
// ---------------------------------------------------------------------------
__global__ void k_gemm_bias_f16(const _Float16* __restrict__ A,
                                const _Float16* __restrict__ W,
                                const float* __restrict__ bias,
                                _Float16* __restrict__ out,
                                int Lrows, int K, int N) {
  const int wid    = blockIdx.x * (blockDim.x >> 5) + (threadIdx.x >> 5);
  const int tilesN = N >> 4;
  const int tiles  = (Lrows >> 4) * tilesN;
  if (wid >= tiles) return;                       // wave-uniform
  const int i0 = (wid / tilesN) << 4;
  const int n0 = (wid % tilesN) << 4;
  const int lane = threadIdx.x & 31;
  const int m = lane & 15, hi = lane >> 4;

  const _Float16* arow = A + (size_t)(i0 + m) * K;   // A row M = lane&15
  const _Float16* brow = W + (size_t)(n0 + m) * K;   // B col N = lane&15
  v8f acc = {};
  for (int kk = 0; kk < K; kk += 32) {
    __builtin_prefetch(arow + kk + 128);
    __builtin_prefetch(brow + kk + 128);
    v16h a = load2x8(arow + kk + hi * 8, arow + kk + 16 + hi * 8);
    v16h b = load2x8(brow + kk + hi * 8, brow + kk + 16 + hi * 8);
    acc = wmma_f16(a, b, acc);
  }
  const int n  = n0 + m;
  const float bv = bias[n];
#pragma unroll
  for (int r = 0; r < 8; ++r) {                      // D: M = r + 8*hi, N = lane&15
    int row = i0 + r + (hi << 3);
    out[(size_t)row * N + n] = (_Float16)(acc[r] + bv);
  }
}

// ---------------------------------------------------------------------------
// Fused masked softmax + mask-dot, flash-style (scores never materialized).
// out[i] += mean_h( sum_j softmax_j(qp_h[i]·kp_h[j]/sqrt(dk), mask) * mask[j] )
// One wave handles (head h, 16 query rows). 8 heads fixed.
// ---------------------------------------------------------------------------
__global__ void k_attn_softmax(const _Float16* __restrict__ qp,
                               const _Float16* __restrict__ kp,
                               const float* __restrict__ maskL,
                               float* __restrict__ out,
                               int L, int C, int dk, float scale) {
  const int rowTiles = L >> 4;
  const int wid = blockIdx.x * (blockDim.x >> 5) + (threadIdx.x >> 5);
  if (wid >= 8 * rowTiles) return;                 // wave-uniform
  const int h  = wid / rowTiles;
  const int i0 = (wid % rowTiles) << 4;
  const int lane = threadIdx.x & 31;
  const int m = lane & 15, hi = lane >> 4;
  const int c0 = h * dk;

  float mrow[8], srow[8], wrow[8];
#pragma unroll
  for (int r = 0; r < 8; ++r) { mrow[r] = -1e30f; srow[r] = 0.f; wrow[r] = 0.f; }

  const _Float16* arow = qp + (size_t)(i0 + m) * C + c0;
  for (int j0 = 0; j0 < L; j0 += 16) {
    const _Float16* brow = kp + (size_t)(j0 + m) * C + c0;
    v8f acc = {};
    for (int kk = 0; kk < dk; kk += 32) {
      v16h a = load2x8(arow + kk + hi * 8, arow + kk + 16 + hi * 8);
      v16h b = load2x8(brow + kk + hi * 8, brow + kk + 16 + hi * 8);
      acc = wmma_f16(a, b, acc);
    }
    const float mj = maskL[j0 + m];                // this lane's key column
    float x[8], tm[8];
#pragma unroll
    for (int r = 0; r < 8; ++r) {
      x[r]  = (mj == 0.f) ? -1e9f : acc[r] * scale;
      tm[r] = x[r];
    }
    // row-max across the 16 lanes holding this D row (xor<=8 stays in half)
    for (int off = 1; off < 16; off <<= 1) {
#pragma unroll
      for (int r = 0; r < 8; ++r) tm[r] = fmaxf(tm[r], __shfl_xor(tm[r], off));
    }
    float es[8], ew[8];
#pragma unroll
    for (int r = 0; r < 8; ++r) {
      float mn   = fmaxf(mrow[r], tm[r]);
      float corr = __expf(mrow[r] - mn);
      float ev   = __expf(x[r] - mn);
      es[r] = ev;
      ew[r] = ev * mj;
      srow[r] *= corr;
      wrow[r] *= corr;
      mrow[r] = mn;
    }
    for (int off = 1; off < 16; off <<= 1) {
#pragma unroll
      for (int r = 0; r < 8; ++r) {
        es[r] += __shfl_xor(es[r], off);
        ew[r] += __shfl_xor(ew[r], off);
      }
    }
#pragma unroll
    for (int r = 0; r < 8; ++r) { srow[r] += es[r]; wrow[r] += ew[r]; }
  }
  if (m == 0) {                                    // lanes 0 and 16 write
#pragma unroll
    for (int r = 0; r < 8; ++r) {
      int row = i0 + r + (hi << 3);
      atomicAdd(&out[row], (wrow[r] / fmaxf(srow[r], 1e-30f)) * 0.125f);
    }
  }
}

// ---------------------------------------------------------------------------
// Conv2d as implicit-im2col WMMA GEMM.
// M = output channels (padded to 16), N = pixels, K = I*ks*ks (padded to 32).
// in: f32 NCHW (converted to f16 in-register), w: f16 (O x I*ks*ks) row-major.
// ---------------------------------------------------------------------------
__global__ void k_conv_wmma(const float* __restrict__ in,
                            const _Float16* __restrict__ w,
                            const float* __restrict__ bias,
                            float* __restrict__ out,
                            int I, int O, int H, int W, int ks, int relu) {
  const int HW  = H * W;
  const int ks2 = ks * ks;
  const int K   = I * ks2;
  const int pad = ks >> 1;
  const int tilesM = (O + 15) >> 4;
  const int tilesN = HW >> 4;
  const int wid = blockIdx.x * (blockDim.x >> 5) + (threadIdx.x >> 5);
  if (wid >= tilesM * tilesN) return;              // wave-uniform
  const int oc0 = (wid % tilesM) << 4;
  const int p0  = (wid / tilesM) << 4;
  const int lane = threadIdx.x & 31;
  const int m = lane & 15, hi = lane >> 4;

  const int oc = oc0 + m;                          // A row for this lane
  const bool wok = (oc < O);
  const _Float16* wrow = w + (size_t)(wok ? oc : 0) * K;

  const int p  = p0 + m;                           // B column (pixel)
  const int py = p / W, px = p % W;

  v8f acc = {};
  for (int kk = 0; kk < K; kk += 32) {
    __builtin_prefetch(wrow + kk + 128);
    v16h a;
#pragma unroll
    for (int t = 0; t < 8; ++t) {
      int k0 = kk + hi * 8 + t;
      int k1 = kk + 16 + hi * 8 + t;
      a[t]     = (wok && k0 < K) ? wrow[k0] : (_Float16)0.f;
      a[t + 8] = (wok && k1 < K) ? wrow[k1] : (_Float16)0.f;
    }
    v16h b;
    const int kb = kk + hi * 8;
#pragma unroll
    for (int t = 0; t < 16; ++t) {
      int k = kb + (t < 8 ? t : 8 + t);
      _Float16 v = (_Float16)0.f;
      if (k < K) {
        int ci  = k / ks2;
        int rem = k - ci * ks2;
        int r   = rem / ks;
        int c   = rem - r * ks;
        int y = py + r - pad;
        int x = px + c - pad;
        if (y >= 0 && y < H && x >= 0 && x < W)
          v = (_Float16)in[(size_t)ci * HW + y * W + x];
      }
      b[t] = v;
    }
    acc = wmma_f16(a, b, acc);
  }
#pragma unroll
  for (int r = 0; r < 8; ++r) {
    int ocr = oc0 + r + (hi << 3);
    if (ocr < O) {
      float v = acc[r] + bias[ocr];
      if (relu) v = fmaxf(v, 0.f);
      out[(size_t)ocr * HW + p] = v;
    }
  }
}

// ---------------------------------------------------------------------------
// GroupNorm(groups=4) + ReLU, in-place. One block per group, LDS reduction.
// ---------------------------------------------------------------------------
__global__ void k_groupnorm_relu(float* __restrict__ x,
                                 const float* __restrict__ gamma,
                                 const float* __restrict__ beta,
                                 int C, int HW) {
  const int g   = blockIdx.x;          // 4 groups
  const int cpg = C >> 2;
  const int n   = cpg * HW;
  float* base = x + (size_t)g * n;
  __shared__ float ssum[256];
  __shared__ float ssq[256];
  float s = 0.f, q = 0.f;
  for (int i = threadIdx.x; i < n; i += blockDim.x) {
    float v = base[i];
    s += v; q += v * v;
  }
  ssum[threadIdx.x] = s;
  ssq[threadIdx.x]  = q;
  __syncthreads();
  for (int off = blockDim.x >> 1; off > 0; off >>= 1) {
    if ((int)threadIdx.x < off) {
      ssum[threadIdx.x] += ssum[threadIdx.x + off];
      ssq[threadIdx.x]  += ssq[threadIdx.x + off];
    }
    __syncthreads();
  }
  float mean = ssum[0] / (float)n;
  float var  = ssq[0] / (float)n - mean * mean;
  float inv  = rsqrtf(var + 1e-5f);
  for (int i = threadIdx.x; i < n; i += blockDim.x) {
    int c = g * cpg + i / HW;
    float v = (base[i] - mean) * inv * gamma[c] + beta[c];
    base[i] = fmaxf(v, 0.f);
  }
}

// ---------------------------------------------------------------------------
// Bilinear align-corners resize (C,Hs,Ws)->(C,oh,ow), optional elementwise add.
// ---------------------------------------------------------------------------
__global__ void k_resize_add(const float* __restrict__ in,
                             const float* __restrict__ add,
                             float* __restrict__ out,
                             int C, int Hs, int Ws, int oh, int ow) {
  int idx = blockIdx.x * blockDim.x + threadIdx.x;
  int total = C * oh * ow;
  if (idx >= total) return;
  int ox = idx % ow;
  int t  = idx / ow;
  int oy = t % oh;
  int c  = t / oh;
  float ys = (oh > 1) ? (float)oy * (float)(Hs - 1) / (float)(oh - 1) : 0.f;
  float xs = (ow > 1) ? (float)ox * (float)(Ws - 1) / (float)(ow - 1) : 0.f;
  int y0 = (int)floorf(ys), x0 = (int)floorf(xs);
  int y1 = min(y0 + 1, Hs - 1), x1 = min(x0 + 1, Ws - 1);
  float wy = ys - (float)y0, wx = xs - (float)x0;
  const float* pc = in + (size_t)c * Hs * Ws;
  float v00 = pc[y0 * Ws + x0], v01 = pc[y0 * Ws + x1];
  float v10 = pc[y1 * Ws + x0], v11 = pc[y1 * Ws + x1];
  float top = v00 * (1.f - wx) + v01 * wx;
  float bot = v10 * (1.f - wx) + v11 * wx;
  float v = top * (1.f - wy) + bot * wy;
  if (add) v += add[idx];
  out[idx] = v;
}

// ===========================================================================
// Host orchestration
// ===========================================================================
extern "C" void kernel_launch(void* const* d_in, const int* in_sizes, int n_in,
                              void* d_out, int out_size, void* d_ws, size_t ws_size,
                              hipStream_t stream) {
  (void)in_sizes; (void)n_in; (void)out_size; (void)ws_size;

  const float* q128  = (const float*)d_in[0];
  const float* q256  = (const float*)d_in[1];
  const float* q512  = (const float*)d_in[2];
  const float* q1024 = (const float*)d_in[3];
  // d_in[4] = s128
  const float* s128  = (const float*)d_in[4];
  const float* s256  = (const float*)d_in[5];
  const float* s512  = (const float*)d_in[6];
  const float* s1024 = (const float*)d_in[7];
  const float* smask = (const float*)d_in[8];

  char* ws = (char*)d_ws;
  size_t off = 0;
  auto alloc = [&](size_t bytes) -> void* {
    void* p = ws + off;
    off = (off + bytes + 255) & ~(size_t)255;
    return p;
  };
  auto conv16 = [&](int idx, int n) -> _Float16* {
    _Float16* d = (_Float16*)alloc((size_t)n * sizeof(_Float16));
    k_f32_to_f16<<<(n + 255) / 256, 256, 0, stream>>>((const float*)d_in[idx], d, n);
    return d;
  };

  // ---- attention weights (f16) + biases (f32) ----
  _Float16* wq0 = conv16(9,  256 * 256);   const float* bq0 = (const float*)d_in[10];
  _Float16* wk0 = conv16(11, 256 * 256);   const float* bk0 = (const float*)d_in[12];
  _Float16* wq1 = conv16(13, 512 * 512);   const float* bq1 = (const float*)d_in[14];
  _Float16* wk1 = conv16(15, 512 * 512);   const float* bk1 = (const float*)d_in[16];
  _Float16* wq2 = conv16(17, 1024 * 1024); const float* bq2 = (const float*)d_in[18];
  _Float16* wk2 = conv16(19, 1024 * 1024); const float* bk2 = (const float*)d_in[20];

  // ---- conv-block params ----
  struct CL { _Float16* w; const float* b; const float* g; const float* bt; int I, O, k; };
  const int cIdx[5]    = {21, 33, 45, 57, 69};
  const int cI[5][3]   = {{2,16,64},{18,16,64},{2,16,64},{128,128,128},{128,128,128}};
  const int cO[5][3]   = {{16,64,128},{16,64,128},{16,64,128},{128,128,128},{128,128,128}};
  const int cK[5][3]   = {{3,3,3},{5,3,3},{5,5,3},{3,3,3},{3,3,3}};
  CL cl[5][3];
  for (int b = 0; b < 5; ++b) {
    for (int l = 0; l < 3; ++l) {
      int base = cIdx[b] + l * 4;
      int I = cI[b][l], O = cO[b][l], k = cK[b][l];
      cl[b][l].w  = conv16(base, O * I * k * k);
      cl[b][l].b  = (const float*)d_in[base + 1];
      cl[b][l].g  = (const float*)d_in[base + 2];
      cl[b][l].bt = (const float*)d_in[base + 3];
      cl[b][l].I = I; cl[b][l].O = O; cl[b][l].k = k;
    }
  }

  // ---- mixer params ----
  struct ML { _Float16* w; const float* b; int I, O; };
  const int mIdx[3]  = {81, 85, 89};
  const int mI[3][2] = {{896,128},{64,64},{16,16}};
  const int mO[3][2] = {{128,64},{64,16},{16,2}};
  ML ml[3][2];
  for (int b = 0; b < 3; ++b) {
    for (int l = 0; l < 2; ++l) {
      int base = mIdx[b] + l * 2;
      int I = mI[b][l], O = mO[b][l];
      ml[b][l].w = conv16(base, O * I * 9);
      ml[b][l].b = (const float*)d_in[base + 1];
      ml[b][l].I = I; ml[b][l].O = O;
    }
  }

  // ---- masks at each scale ----
  float* m48 = (float*)alloc(48 * 48 * 4);
  float* m24 = (float*)alloc(24 * 24 * 4);
  float* m12 = (float*)alloc(12 * 12 * 4);
  k_resize_add<<<(48*48 + 255)/256, 256, 0, stream>>>(smask, nullptr, m48, 1, 384, 384, 48, 48);
  k_resize_add<<<(24*24 + 255)/256, 256, 0, stream>>>(smask, nullptr, m24, 1, 384, 384, 24, 24);
  k_resize_add<<<(12*12 + 255)/256, 256, 0, stream>>>(smask, nullptr, m12, 1, 384, 384, 12, 12);

  // ---- attention scratch (max L*C = 2304*256) ----
  const size_t maxLC = 2304 * 256;
  _Float16* Qb = (_Float16*)alloc(maxLC * 2);
  _Float16* Sb = (_Float16*)alloc(maxLC * 2);
  _Float16* QP = (_Float16*)alloc(maxLC * 2);
  _Float16* KP = (_Float16*)alloc(maxLC * 2);

  float* coarse1 = (float*)alloc(2  * 144  * 4);
  float* coarse2 = (float*)alloc(18 * 576  * 4);
  float* coarse3 = (float*)alloc(2  * 2304 * 4);
  hipMemsetAsync(coarse1, 0, 2  * 144  * 4, stream);
  hipMemsetAsync(coarse2, 0, 18 * 576  * 4, stream);
  hipMemsetAsync(coarse3, 0, 2  * 2304 * 4, stream);

  auto run_attn = [&](const float* qf, const float* sf, int C, int L,
                      _Float16* wq, const float* bq, _Float16* wk, const float* bk,
                      const float* maskL, float* outPlane) {
    int n = L * C;
    k_prep_qs<<<(n + 255) / 256, 256, 0, stream>>>(qf, Qb, C, L);
    k_prep_qs<<<(n + 255) / 256, 256, 0, stream>>>(sf, Sb, C, L);
    int tiles = (L / 16) * (C / 16);
    k_gemm_bias_f16<<<(tiles + 3) / 4, 128, 0, stream>>>(Qb, wq, bq, QP, L, C, C);
    k_gemm_bias_f16<<<(tiles + 3) / 4, 128, 0, stream>>>(Sb, wk, bk, KP, L, C, C);
    int dk = C / 8;
    int waves = 8 * (L / 16);
    float scale = 1.f / sqrtf((float)dk);
    k_attn_softmax<<<(waves + 3) / 4, 128, 0, stream>>>(QP, KP, maskL, outPlane, L, C, dk, scale);
  };

  for (int i = 0; i < 2; ++i)
    run_attn(q256 + (size_t)i * 256 * 2304, s256 + (size_t)i * 256 * 2304,
             256, 2304, wq0, bq0, wk0, bk0, m48, coarse3 + (size_t)i * 2304);
  for (int i = 0; i < 18; ++i)
    run_attn(q512 + (size_t)i * 512 * 576, s512 + (size_t)i * 512 * 576,
             512, 576, wq1, bq1, wk1, bk1, m24, coarse2 + (size_t)i * 576);
  for (int i = 0; i < 2; ++i)
    run_attn(q1024 + (size_t)i * 1024 * 144, s1024 + (size_t)i * 1024 * 144,
             1024, 144, wq2, bq2, wk2, bk2, m12, coarse1 + (size_t)i * 144);

  // ---- conv blocks (conv -> GN -> ReLU, x3) ----
  float* tA = (float*)alloc((size_t)128 * 2304 * 4);
  float* tB = (float*)alloc((size_t)128 * 2304 * 4);
  auto run_conv_gn = [&](const float* in, float* out, const CL& L, int H, int W) {
    int HW = H * W;
    int tiles = ((L.O + 15) / 16) * (HW / 16);
    k_conv_wmma<<<(tiles + 3) / 4, 128, 0, stream>>>(in, L.w, L.b, out, L.I, L.O, H, W, L.k, 0);
    k_groupnorm_relu<<<4, 256, 0, stream>>>(out, L.g, L.bt, L.O, HW);
  };
  auto run_block = [&](const float* in, float* out, CL* Ls, int H, int W) {
    run_conv_gn(in, tA, Ls[0], H, W);
    run_conv_gn(tA, tB, Ls[1], H, W);
    run_conv_gn(tB, out, Ls[2], H, W);
  };

  float* c1 = (float*)alloc((size_t)128 * 144  * 4);
  float* c2 = (float*)alloc((size_t)128 * 576  * 4);
  float* c3 = (float*)alloc((size_t)128 * 2304 * 4);
  run_block(coarse1, c1, cl[0], 12, 12);
  run_block(coarse2, c2, cl[1], 24, 24);
  run_block(coarse3, c3, cl[2], 48, 48);

  float* mixA = (float*)alloc((size_t)128 * 576 * 4);
  k_resize_add<<<(128*576 + 255)/256, 256, 0, stream>>>(c1, c2, mixA, 128, 12, 12, 24, 24);
  float* mixB = (float*)alloc((size_t)128 * 576 * 4);
  run_block(mixA, mixB, cl[3], 24, 24);
  float* mixC = (float*)alloc((size_t)128 * 2304 * 4);
  k_resize_add<<<(128*2304 + 255)/256, 256, 0, stream>>>(mixB, c3, mixC, 128, 24, 24, 48, 48);
  float* mixD = (float*)alloc((size_t)128 * 2304 * 4);
  run_block(mixC, mixD, cl[4], 48, 48);

  // ---- concat -> 896ch @ 96x96 ----
  const size_t P96 = 9216;   // 96*96
  float* B96 = (float*)alloc((size_t)896 * P96 * 4);
  k_resize_add<<<((int)(128*P96) + 255)/256, 256, 0, stream>>>(mixD, nullptr, B96, 128, 48, 48, 96, 96);
  k_resize_add<<<((int)(256*P96) + 255)/256, 256, 0, stream>>>(q256 + (size_t)256*2304, nullptr,
                                                               B96 + 128*P96, 256, 48, 48, 96, 96);
  k_resize_add<<<((int)(256*P96) + 255)/256, 256, 0, stream>>>(s256 + (size_t)256*2304, nullptr,
                                                               B96 + 384*P96, 256, 48, 48, 96, 96);
  hipMemcpyAsync(B96 + 640*P96, q128 + (size_t)128*P96, (size_t)128*P96*4, hipMemcpyDeviceToDevice, stream);
  hipMemcpyAsync(B96 + 768*P96, s128 + (size_t)128*P96, (size_t)128*P96*4, hipMemcpyDeviceToDevice, stream);

  // ---- mixers ----
  auto run_mix = [&](const float* in, float* out, const ML& L, int H, int W, int relu) {
    int HW = H * W;
    int tiles = ((L.O + 15) / 16) * (HW / 16);
    k_conv_wmma<<<(tiles + 3) / 4, 128, 0, stream>>>(in, L.w, L.b, out, L.I, L.O, H, W, 3, relu);
  };
  float* M1a = (float*)alloc((size_t)128 * P96 * 4);
  run_mix(B96, M1a, ml[0][0], 96, 96, 1);
  float* M1b = (float*)alloc((size_t)64 * P96 * 4);
  run_mix(M1a, M1b, ml[0][1], 96, 96, 1);

  const size_t P192 = 36864;  // 192*192
  float* R192 = (float*)alloc((size_t)64 * P192 * 4);
  k_resize_add<<<((int)(64*P192) + 255)/256, 256, 0, stream>>>(M1b, nullptr, R192, 64, 96, 96, 192, 192);
  float* M2a = (float*)alloc((size_t)64 * P192 * 4);
  run_mix(R192, M2a, ml[1][0], 192, 192, 1);
  float* M2b = (float*)alloc((size_t)16 * P192 * 4);
  run_mix(M2a, M2b, ml[1][1], 192, 192, 1);

  const size_t P384 = 147456; // 384*384
  float* R384 = (float*)alloc((size_t)16 * P384 * 4);
  k_resize_add<<<((int)(16*P384) + 255)/256, 256, 0, stream>>>(M2b, nullptr, R384, 16, 192, 192, 384, 384);
  float* M3a = (float*)alloc((size_t)16 * P384 * 4);
  run_mix(R384, M3a, ml[2][0], 384, 384, 1);
  run_mix(M3a, (float*)d_out, ml[2][1], 384, 384, 0);  // (1,2,384,384)
}
